// ClofNetPyG_77618648973488
// MI455X (gfx1250) — matrix operand order repaired and checked by reading, not compiled, not run on hardware
//
#include <hip/hip_runtime.h>
#include <hip/hip_bf16.h>
#include <math.h>

// ClofNet forward for MI455X (gfx1250, wave32).
// All dense GEMMs use v_wmma_f32_16x16x32_bf16 (bf16 in, fp32 acc):
//   edge MLP  : [E x 352(pad/permuted 321)] x [352 x 128] -> silu -> x [128 x 128] -> silu
//   node MLP  : [N x 256] x [256 x 128] -> silu -> x [128 x 128]
//   coord MLP : [E x 128] x [128 x 128] -> silu -> (x [128 x 3] via VALU dot)
// Weights pre-packed bf16, K-pair interleaved AND fragment-major per K=32 chunk:
//   chunk ck, column n, pair pl (0..15):  word at ck*2048 + n*16 + pl
// so a lane's B-fragment (pairs hh*4..hh*4+3 and 8+hh*4..8+hh*4+3 of column n) is 2x ds_load_b128.
// A-fragment: lane m=L&15, h=L>>4; VGPR 0..3 <- K=8h+0..7, VGPR 4..7 <- K=16+8h+0..7
// (two 16B-aligned runs -> 2x ds_load_b128). Weight chunks double-buffered in LDS.
// edge-MLP K order is permuted (radial moved to k=320) so A staging is pure u32 copies;
// the e1 weight pack applies the same permutation.

#define HID 128
#define FEPS 1e-8f

typedef __attribute__((ext_vector_type(16))) __bf16 v16bf;
typedef __attribute__((ext_vector_type(8)))  float  v8f;

union AFrag { unsigned u[8]; v16bf v; };

__device__ __forceinline__ unsigned short f2bf(float f) {
    unsigned u = __float_as_uint(f);
    unsigned r = ((u >> 16) & 1u) + 0x7FFFu;      // round to nearest even
    return (unsigned short)((u + r) >> 16);
}
__device__ __forceinline__ float bf2f(unsigned short b) {
    return __uint_as_float(((unsigned)b) << 16);
}
__device__ __forceinline__ float silu_f(float x) { return x / (1.0f + __expf(-x)); }
__device__ __forceinline__ unsigned pack2(unsigned short lo, unsigned short hi) {
    return (unsigned)lo | ((unsigned)hi << 16);
}

__device__ __forceinline__ void load_afrag_lds(AFrag& a, const unsigned short* ar, int hh) {
    *(uint4*)&a.u[0] = *(const uint4*)(ar + hh * 8);
    *(uint4*)&a.u[4] = *(const uint4*)(ar + 16 + hh * 8);
}
__device__ __forceinline__ void load_bfrag(AFrag& b, const unsigned* Wc, int n, int hh) {
    const unsigned* bp = Wc + n * 16 + hh * 4;
    *(uint4*)&b.u[0] = *(const uint4*)bp;
    *(uint4*)&b.u[4] = *(const uint4*)(bp + 8);
}
__device__ __forceinline__ void stage_chunk(unsigned* dst, const unsigned* src, int tid) {
    const uint4* s4 = (const uint4*)src;
    uint4* d4 = (uint4*)dst;
#pragma unroll
    for (int i = 0; i < 4; ++i) d4[tid + i * 128] = s4[tid + i * 128];
}

// ---------------------------------------------------------------- weights ---
// generic: K-pair interleaved, fragment-major per chunk
__global__ void pack_weights_kernel(const float* __restrict__ W, unsigned* __restrict__ out,
                                    int K, int Kpad, int Nc) {
    int t = blockIdx.x * blockDim.x + threadIdx.x;
    int total = (Kpad >> 1) * Nc;
    if (t >= total) return;
    int p = t / Nc, n = t - p * Nc;
    int k0 = 2 * p;
    unsigned short lo = (k0     < K) ? f2bf(W[(size_t)k0 * Nc + n])       : (unsigned short)0;
    unsigned short hi = (k0 + 1 < K) ? f2bf(W[(size_t)(k0 + 1) * Nc + n]) : (unsigned short)0;
    out[(p >> 4) * (16 * Nc) + n * 16 + (p & 15)] = pack2(lo, hi);
}

// e1 with K permutation: ourK<256 -> srcK=ourK ; 256..319 -> eattr rows 257..320 ;
// 320 -> radial row 256 ; >=321 -> zero pad.  (W is 321 x 128)
__device__ __forceinline__ int e1_src_row(int k) {
    if (k < 256) return k;
    if (k < 320) return 257 + (k - 256);
    if (k == 320) return 256;
    return -1;
}
__global__ void pack_weights_e1_kernel(const float* __restrict__ W, unsigned* __restrict__ out) {
    const int Nc = 128;
    int t = blockIdx.x * blockDim.x + threadIdx.x;
    int total = 176 * Nc;                       // Kpad = 352
    if (t >= total) return;
    int p = t / Nc, n = t - p * Nc;
    int s0 = e1_src_row(2 * p), s1 = e1_src_row(2 * p + 1);
    unsigned short lo = (s0 >= 0) ? f2bf(W[(size_t)s0 * Nc + n]) : (unsigned short)0;
    unsigned short hi = (s1 >= 0) ? f2bf(W[(size_t)s1 * Nc + n]) : (unsigned short)0;
    out[(p >> 4) * 2048 + n * 16 + (p & 15)] = pack2(lo, hi);
}

// ------------------------------------------------------------------ embed ---
__global__ void emb_in_kernel(const float* __restrict__ hin, const float* __restrict__ W,
                              const float* __restrict__ b, float* __restrict__ h,
                              unsigned short* __restrict__ h_bf, int N) {
    int t = blockIdx.x * blockDim.x + threadIdx.x;
    if (t >= N * HID) return;
    int node = t / HID, c = t - node * HID;
    float s = b[c];
#pragma unroll
    for (int k = 0; k < 16; ++k) s += hin[node * 16 + k] * W[k * HID + c];
    h[t] = s;
    h_bf[t] = f2bf(s);
}

__global__ void emb_out_kernel(const float* __restrict__ h, const float* __restrict__ W,
                               const float* __restrict__ b, float* __restrict__ out, int N) {
    int node = blockIdx.x * blockDim.x + threadIdx.x;
    if (node >= N) return;
    float acc[16];
#pragma unroll
    for (int j = 0; j < 16; ++j) acc[j] = b[j];
    const float* r = &h[(size_t)node * HID];
    for (int k = 0; k < HID; ++k) {
        float v = r[k];
#pragma unroll
        for (int j = 0; j < 16; ++j) acc[j] += v * W[k * 16 + j];
    }
#pragma unroll
    for (int j = 0; j < 16; ++j) out[(size_t)node * 16 + j] = acc[j];
}

// --------------------------------------------------------------- geometry ---
__global__ void zero_kernel(float* __restrict__ p, size_t n) {
    size_t t = (size_t)blockIdx.x * blockDim.x + threadIdx.x;
    if (t < n) p[t] = 0.0f;
}

__global__ void centroid_kernel(const float* __restrict__ x, const int* __restrict__ nn_ptr,
                                float* __restrict__ csum, int N) {
    int t = blockIdx.x * blockDim.x + threadIdx.x;
    if (t >= N * 3) return;
    int node = t / 3, d = t - node * 3;
    int nn = nn_ptr[0];
    atomicAdd(&csum[(node / nn) * 3 + d], x[t]);
}

__global__ void coord_init_kernel(const float* __restrict__ x, const int* __restrict__ nn_ptr,
                                  const float* __restrict__ csum, float* __restrict__ coord, int N) {
    int t = blockIdx.x * blockDim.x + threadIdx.x;
    if (t >= N * 3) return;
    int node = t / 3, d = t - node * 3;
    int nn = nn_ptr[0];
    coord[t] = x[t] - csum[(node / nn) * 3 + d] / (float)nn;
}

__global__ void xout_kernel(const float* __restrict__ coord, const float* __restrict__ csum,
                            const int* __restrict__ nn_ptr, float* __restrict__ out, int N) {
    int t = blockIdx.x * blockDim.x + threadIdx.x;
    if (t >= N * 3) return;
    int node = t / 3, d = t - node * 3;
    int nn = nn_ptr[0];
    out[t] = coord[t] + csum[(node / nn) * 3 + d] / (float)nn;
}

// geom per edge: [radial, cd(3), cc(3), cv(3)]
__global__ void frame_kernel(const int* __restrict__ row, const int* __restrict__ col,
                             const float* __restrict__ coord, float* __restrict__ geom, int E) {
    int e = blockIdx.x * blockDim.x + threadIdx.x;
    if (e >= E) return;
    int ri = row[e], ci = col[e];
    float ax = coord[ri * 3 + 0], ay = coord[ri * 3 + 1], az = coord[ri * 3 + 2];
    float bx = coord[ci * 3 + 0], by = coord[ci * 3 + 1], bz = coord[ci * 3 + 2];
    float dx = ax - bx, dy = ay - by, dz = az - bz;
    float radial = dx * dx + dy * dy + dz * dz;
    float cx = ay * bz - az * by, cy = az * bx - ax * bz, cz = ax * by - ay * bx;
    float idn = 1.0f / (sqrtf(radial + FEPS) + 1.0f);
    dx *= idn; dy *= idn; dz *= idn;
    float icn = 1.0f / (sqrtf(cx * cx + cy * cy + cz * cz + FEPS) + 1.0f);
    cx *= icn; cy *= icn; cz *= icn;
    float vx = dy * cz - dz * cy, vy = dz * cx - dx * cz, vz = dx * cy - dy * cx;
    float* g = &geom[(size_t)e * 10];
    g[0] = radial;
    g[1] = dx; g[2] = dy; g[3] = dz;
    g[4] = cx; g[5] = cy; g[6] = cz;
    g[7] = vx; g[8] = vy; g[9] = vz;
}

// -------------------------------------------------- initial edge features ---
__global__ __launch_bounds__(128) void edge_attr_kernel(
        const int* __restrict__ row, const int* __restrict__ col,
        const float* __restrict__ coord, const float* __restrict__ geom,
        const float* __restrict__ f1W, const float* __restrict__ f1b,
        const float* __restrict__ f2W, const float* __restrict__ f2b,
        unsigned short* __restrict__ eattr, int E) {
    __shared__ float sW1[8 * 64];
    __shared__ float sb1[64];
    __shared__ float sW2[64 * 64];
    __shared__ float sb2[64];
    for (int i = threadIdx.x; i < 8 * 64; i += blockDim.x) sW1[i] = f1W[i];
    for (int i = threadIdx.x; i < 64; i += blockDim.x) { sb1[i] = f1b[i]; sb2[i] = f2b[i]; }
    for (int i = threadIdx.x; i < 64 * 64; i += blockDim.x) sW2[i] = f2W[i];
    __syncthreads();
    int e = blockIdx.x * blockDim.x + threadIdx.x;
    if (e >= E) return;
    const float* g = &geom[(size_t)e * 10];
    int ri = row[e], ci = col[e];
    float rx = coord[ri * 3 + 0], ry = coord[ri * 3 + 1], rz = coord[ri * 3 + 2];
    float qx = coord[ci * 3 + 0], qy = coord[ci * 3 + 1], qz = coord[ci * 3 + 2];
    float ci0 = g[1] * rx + g[2] * ry + g[3] * rz;
    float ci1 = g[4] * rx + g[5] * ry + g[6] * rz;
    float ci2 = g[7] * rx + g[8] * ry + g[9] * rz;
    float cj0 = g[1] * qx + g[2] * qy + g[3] * qz;
    float cj1 = g[4] * qx + g[5] * qy + g[6] * qz;
    float cj2 = g[7] * qx + g[8] * qy + g[9] * qz;
    float ni = sqrtf(ci0 * ci0 + ci1 * ci1 + ci2 * ci2) + 1e-5f;
    float nj = sqrtf(cj0 * cj0 + cj1 * cj1 + cj2 * cj2) + 1e-5f;
    float cosv = (ci0 * cj0 + ci1 * cj1 + ci2 * cj2) / (ni * nj);
    float s2 = 1.0f - cosv * cosv;
    if (s2 < 1e-12f) s2 = 1e-12f;
    float sinv = sqrtf(s2);
    float feat[8] = { sinv, cosv, ci0, ci1, ci2, cj0, cj1, cj2 };
    float t1[64];
#pragma unroll 4
    for (int j = 0; j < 64; ++j) {
        float s = sb1[j];
#pragma unroll
        for (int k = 0; k < 8; ++k) s += feat[k] * sW1[k * 64 + j];
        t1[j] = silu_f(s);
    }
#pragma unroll 2
    for (int j = 0; j < 64; ++j) {
        float s = sb2[j];
        for (int k = 0; k < 64; ++k) s += t1[k] * sW2[k * 64 + j];
        eattr[(size_t)e * 64 + j] = f2bf(silu_f(s));
    }
}

// --------------------------------------------------------------- edge MLP ---
__global__ __launch_bounds__(128) void edge_mlp_kernel(
        const int* __restrict__ row, const int* __restrict__ col,
        const unsigned short* __restrict__ h_bf,
        const unsigned short* __restrict__ eattr,
        const float* __restrict__ geom,
        const unsigned* __restrict__ w1p, const float* __restrict__ b1,
        const unsigned* __restrict__ w2p, const float* __restrict__ b2,
        unsigned short* __restrict__ ef, int E) {
    constexpr int KP = 352;                    // 321 padded/permuted to 11 * 32
    __shared__ unsigned short Abuf[64 * KP];   // 45056 B
    __shared__ unsigned Wch[2][2048];          // 2 x 8192 B, double buffered
    unsigned short* T1 = Abuf;                 // [64][128], aliased after GEMM1

    const int tid  = threadIdx.x;
    const int w    = tid >> 5;
    const int lane = tid & 31;
    const int m    = lane & 15;
    const int hh   = lane >> 4;
    const int e0   = blockIdx.x * 64 + w * 16;

    // stage A rows as u32 words: [h[col] 0..63 | h[row] 64..127 | eattr 128..159 |
    //                             {radial,0} 160 | zeros 161..175]
    for (int r = 0; r < 16; ++r) {
        int e = e0 + r; if (e >= E) e = E - 1;
        int ci = col[e], ri = row[e];
        unsigned radw = (unsigned)f2bf(geom[(size_t)e * 10]);
        unsigned* dst = (unsigned*)&Abuf[(w * 16 + r) * KP];
        const unsigned* hc = (const unsigned*)h_bf + (size_t)ci * 64;
        const unsigned* hr = (const unsigned*)h_bf + (size_t)ri * 64;
        const unsigned* ea = (const unsigned*)eattr + (size_t)e * 32;
        for (int cw = lane; cw < 176; cw += 32) {
            unsigned v;
            if (cw < 64)        v = hc[cw];
            else if (cw < 128)  v = hr[cw - 64];
            else if (cw < 160)  v = ea[cw - 128];
            else if (cw == 160) v = radw;
            else                v = 0u;
            dst[cw] = v;
        }
    }

    v8f acc[8];
#pragma unroll
    for (int i = 0; i < 8; ++i)
#pragma unroll
        for (int j = 0; j < 8; ++j) acc[i][j] = 0.0f;

    stage_chunk(Wch[0], w1p, tid);
    __syncthreads();
    for (int kt = 0; kt < 11; ++kt) {
        int cur = kt & 1;
        if (kt + 1 < 11) stage_chunk(Wch[cur ^ 1], w1p + (kt + 1) * 2048, tid);
        AFrag a;
        load_afrag_lds(a, &Abuf[(w * 16 + m) * KP + kt * 32], hh);
#pragma unroll
        for (int nt = 0; nt < 8; ++nt) {
            AFrag b;
            load_bfrag(b, Wch[cur], nt * 16 + m, hh);
            acc[nt] = __builtin_amdgcn_wmma_f32_16x16x32_bf16(
                false, a.v, false, b.v, (short)0, acc[nt], false, false);
        }
        __syncthreads();
    }

    // silu+bias -> T1 (aliases Abuf; all Abuf reads retired by the final barrier)
#pragma unroll
    for (int nt = 0; nt < 8; ++nt)
#pragma unroll
        for (int r = 0; r < 8; ++r) {
            int mm = r + 8 * hh;
            float xv = acc[nt][r] + b1[nt * 16 + m];
            T1[(w * 16 + mm) * HID + nt * 16 + m] = f2bf(silu_f(xv));
        }

    v8f acc2[8];
#pragma unroll
    for (int i = 0; i < 8; ++i)
#pragma unroll
        for (int j = 0; j < 8; ++j) acc2[i][j] = 0.0f;

    stage_chunk(Wch[0], w2p, tid);
    __syncthreads();
    for (int kt = 0; kt < 4; ++kt) {
        int cur = kt & 1;
        if (kt + 1 < 4) stage_chunk(Wch[cur ^ 1], w2p + (kt + 1) * 2048, tid);
        AFrag a;
        load_afrag_lds(a, &T1[(w * 16 + m) * HID + kt * 32], hh);
#pragma unroll
        for (int nt = 0; nt < 8; ++nt) {
            AFrag b;
            load_bfrag(b, Wch[cur], nt * 16 + m, hh);
            acc2[nt] = __builtin_amdgcn_wmma_f32_16x16x32_bf16(
                false, a.v, false, b.v, (short)0, acc2[nt], false, false);
        }
        __syncthreads();
    }

#pragma unroll
    for (int nt = 0; nt < 8; ++nt)
#pragma unroll
        for (int r = 0; r < 8; ++r) {
            int mm = r + 8 * hh;
            int e = e0 + mm;
            if (e < E) {
                float xv = acc2[nt][r] + b2[nt * 16 + m];
                ef[(size_t)e * HID + nt * 16 + m] = f2bf(silu_f(xv));
            }
        }
}

// ----------------------------------------------------- segment sum (agg) ----
__global__ void agg_kernel(const int* __restrict__ col, const unsigned short* __restrict__ ef,
                           float* __restrict__ agg, int E) {
    int t = blockIdx.x * blockDim.x + threadIdx.x;
    if (t >= E * 32) return;
    int e = t >> 5, q = t & 31;
    int ci = col[e];
    const unsigned* src = (const unsigned*)(ef + (size_t)e * HID) + q * 2;
    unsigned a0 = src[0], a1 = src[1];
    float* dst = &agg[(size_t)ci * HID + q * 4];
    atomicAdd(dst + 0, bf2f((unsigned short)(a0 & 0xFFFFu)));
    atomicAdd(dst + 1, bf2f((unsigned short)(a0 >> 16)));
    atomicAdd(dst + 2, bf2f((unsigned short)(a1 & 0xFFFFu)));
    atomicAdd(dst + 3, bf2f((unsigned short)(a1 >> 16)));
}

// --------------------------------------------------------------- node MLP ---
__global__ __launch_bounds__(128) void node_mlp_kernel(
        const unsigned short* __restrict__ h_bf, const float* __restrict__ agg,
        const unsigned* __restrict__ w1p, const float* __restrict__ b1,
        const unsigned* __restrict__ w2p, const float* __restrict__ b2,
        float* __restrict__ hn, int N) {
    __shared__ unsigned short T1[64 * HID];
    __shared__ unsigned Wch[2][2048];
    const int tid  = threadIdx.x;
    const int w    = tid >> 5;
    const int lane = tid & 31;
    const int m    = lane & 15;
    const int hh   = lane >> 4;
    const int n0   = blockIdx.x * 64 + w * 16;
    int nodeRow = n0 + m; if (nodeRow >= N) nodeRow = N - 1;

    v8f acc[8];
#pragma unroll
    for (int i = 0; i < 8; ++i)
#pragma unroll
        for (int j = 0; j < 8; ++j) acc[i][j] = 0.0f;

    stage_chunk(Wch[0], w1p, tid);
    __syncthreads();
    for (int kt = 0; kt < 8; ++kt) {           // K = 256 : [h | agg]
        int cur = kt & 1;
        if (kt + 1 < 8) stage_chunk(Wch[cur ^ 1], w1p + (kt + 1) * 2048, tid);
        AFrag a;
        if (kt < 4) {
            const unsigned short* ar = &h_bf[(size_t)nodeRow * HID + kt * 32];
            *(uint4*)&a.u[0] = *(const uint4*)(ar + hh * 8);
            *(uint4*)&a.u[4] = *(const uint4*)(ar + 16 + hh * 8);
        } else {
            const float* ar = &agg[(size_t)nodeRow * HID + (kt - 4) * 32];
            float4 f0 = *(const float4*)(ar + hh * 8);
            float4 f1 = *(const float4*)(ar + hh * 8 + 4);
            float4 f2 = *(const float4*)(ar + 16 + hh * 8);
            float4 f3 = *(const float4*)(ar + 16 + hh * 8 + 4);
            a.u[0] = pack2(f2bf(f0.x), f2bf(f0.y));
            a.u[1] = pack2(f2bf(f0.z), f2bf(f0.w));
            a.u[2] = pack2(f2bf(f1.x), f2bf(f1.y));
            a.u[3] = pack2(f2bf(f1.z), f2bf(f1.w));
            a.u[4] = pack2(f2bf(f2.x), f2bf(f2.y));
            a.u[5] = pack2(f2bf(f2.z), f2bf(f2.w));
            a.u[6] = pack2(f2bf(f3.x), f2bf(f3.y));
            a.u[7] = pack2(f2bf(f3.z), f2bf(f3.w));
        }
#pragma unroll
        for (int nt = 0; nt < 8; ++nt) {
            AFrag b;
            load_bfrag(b, Wch[cur], nt * 16 + m, hh);
            acc[nt] = __builtin_amdgcn_wmma_f32_16x16x32_bf16(
                false, a.v, false, b.v, (short)0, acc[nt], false, false);
        }
        __syncthreads();
    }

#pragma unroll
    for (int nt = 0; nt < 8; ++nt)
#pragma unroll
        for (int r = 0; r < 8; ++r) {
            int mm = r + 8 * hh;
            float xv = acc[nt][r] + b1[nt * 16 + m];
            T1[(w * 16 + mm) * HID + nt * 16 + m] = f2bf(silu_f(xv));
        }

    v8f acc2[8];
#pragma unroll
    for (int i = 0; i < 8; ++i)
#pragma unroll
        for (int j = 0; j < 8; ++j) acc2[i][j] = 0.0f;

    stage_chunk(Wch[0], w2p, tid);
    __syncthreads();
    for (int kt = 0; kt < 4; ++kt) {
        int cur = kt & 1;
        if (kt + 1 < 4) stage_chunk(Wch[cur ^ 1], w2p + (kt + 1) * 2048, tid);
        AFrag a;
        load_afrag_lds(a, &T1[(w * 16 + m) * HID + kt * 32], hh);
#pragma unroll
        for (int nt = 0; nt < 8; ++nt) {
            AFrag b;
            load_bfrag(b, Wch[cur], nt * 16 + m, hh);
            acc2[nt] = __builtin_amdgcn_wmma_f32_16x16x32_bf16(
                false, a.v, false, b.v, (short)0, acc2[nt], false, false);
        }
        __syncthreads();
    }

#pragma unroll
    for (int nt = 0; nt < 8; ++nt)
#pragma unroll
        for (int r = 0; r < 8; ++r) {
            int mm = r + 8 * hh;
            int node = n0 + mm;
            if (node < N)
                hn[(size_t)node * HID + nt * 16 + m] = acc2[nt][r] + b2[nt * 16 + m];
        }
}

// -------------------------------------------------------------- layernorm ---
__global__ void layernorm_kernel(const float* __restrict__ hn, const float* __restrict__ g,
                                 const float* __restrict__ b, float* __restrict__ h,
                                 unsigned short* __restrict__ h_bf, int N) {
    int node = blockIdx.x * blockDim.x + threadIdx.x;
    if (node >= N) return;
    const float* r = &hn[(size_t)node * HID];
    float s = 0.0f, ss = 0.0f;
    for (int k = 0; k < HID; ++k) { float v = r[k]; s += v; ss += v * v; }
    float mu = s * (1.0f / HID);
    float var = ss * (1.0f / HID) - mu * mu;
    if (var < 0.0f) var = 0.0f;
    float inv = rsqrtf(var + 1e-5f);
    for (int k = 0; k < HID; ++k) {
        float v = (r[k] - mu) * inv * g[k] + b[k];
        h[(size_t)node * HID + k] = v;
        h_bf[(size_t)node * HID + k] = f2bf(v);
    }
}

// --------------------------------------- coord MLP (c1 WMMA + c2 + scatter) --
__global__ __launch_bounds__(128) void coord_update_kernel(
        const int* __restrict__ row, const unsigned short* __restrict__ ef,
        const unsigned* __restrict__ wc1p, const float* __restrict__ bc1,
        const float* __restrict__ c2W, const float* __restrict__ c2b,
        const float* __restrict__ geom, float* __restrict__ coord, int E) {
    __shared__ unsigned short T1[64 * HID];
    __shared__ unsigned Wch[2][2048];
    const int tid  = threadIdx.x;
    const int w    = tid >> 5;
    const int lane = tid & 31;
    const int m    = lane & 15;
    const int hh   = lane >> 4;
    const int e0   = blockIdx.x * 64 + w * 16;
    int eRow = e0 + m; if (eRow >= E) eRow = E - 1;

    v8f acc[8];
#pragma unroll
    for (int i = 0; i < 8; ++i)
#pragma unroll
        for (int j = 0; j < 8; ++j) acc[i][j] = 0.0f;

    stage_chunk(Wch[0], wc1p, tid);
    __syncthreads();
    for (int kt = 0; kt < 4; ++kt) {
        int cur = kt & 1;
        if (kt + 1 < 4) stage_chunk(Wch[cur ^ 1], wc1p + (kt + 1) * 2048, tid);
        AFrag a;
        const unsigned short* ar = &ef[(size_t)eRow * HID + kt * 32];
        *(uint4*)&a.u[0] = *(const uint4*)(ar + hh * 8);
        *(uint4*)&a.u[4] = *(const uint4*)(ar + 16 + hh * 8);
#pragma unroll
        for (int nt = 0; nt < 8; ++nt) {
            AFrag b;
            load_bfrag(b, Wch[cur], nt * 16 + m, hh);
            acc[nt] = __builtin_amdgcn_wmma_f32_16x16x32_bf16(
                false, a.v, false, b.v, (short)0, acc[nt], false, false);
        }
        __syncthreads();
    }

#pragma unroll
    for (int nt = 0; nt < 8; ++nt)
#pragma unroll
        for (int r = 0; r < 8; ++r) {
            int mm = r + 8 * hh;
            float xv = acc[nt][r] + bc1[nt * 16 + m];
            T1[(w * 16 + mm) * HID + nt * 16 + m] = f2bf(silu_f(xv));
        }
    __syncthreads();

    if (lane < 16) {
        int e = e0 + lane;
        if (e < E) {
            float s0 = 0.0f, s1 = 0.0f, s2 = 0.0f;
            const unsigned short* t = &T1[(w * 16 + lane) * HID];
            for (int k = 0; k < HID; ++k) {
                float tv = bf2f(t[k]);
                s0 += tv * c2W[k * 3 + 0];
                s1 += tv * c2W[k * 3 + 1];
                s2 += tv * c2W[k * 3 + 2];
            }
            s0 += c2b[0]; s1 += c2b[1]; s2 += c2b[2];
            const float* g = &geom[(size_t)e * 10];
            int ri = row[e];
#pragma unroll
            for (int d = 0; d < 3; ++d) {
                float tr = g[1 + d] * s0 + g[4 + d] * s1 + g[7 + d] * s2;
                atomicAdd(&coord[ri * 3 + d], tr);
            }
        }
    }
}

// ------------------------------------------------------------------- host ---
extern "C" void kernel_launch(void* const* d_in, const int* in_sizes, int n_in,
                              void* d_out, int out_size, void* d_ws, size_t ws_size,
                              hipStream_t stream) {
    const int N = in_sizes[0] / 16;
    const int E = in_sizes[1] / 2;

    const float* h_in = (const float*)d_in[0];
    const int*   eidx = (const int*)d_in[1];
    const int*   row  = eidx;
    const int*   col  = eidx + E;
    const float* x       = (const float*)d_in[2];
    const int*   nn_ptr  = (const int*)d_in[3];
    const float* embin_W = (const float*)d_in[4];
    const float* embin_b = (const float*)d_in[5];
    const float* embout_W = (const float*)d_in[6];
    const float* embout_b = (const float*)d_in[7];
    const float* f1W = (const float*)d_in[8];
    const float* f1b = (const float*)d_in[9];
    const float* f2W = (const float*)d_in[10];
    const float* f2b = (const float*)d_in[11];

    size_t off = 0;
    auto alloc = [&](size_t bytes) -> char* {
        char* p = (char*)d_ws + off;
        off += (bytes + 255) & ~(size_t)255;
        return p;
    };
    float*          h     = (float*)alloc((size_t)N * HID * 4);
    unsigned short* h_bf  = (unsigned short*)alloc((size_t)N * HID * 2);
    float*          aggb  = (float*)alloc((size_t)N * HID * 4);
    float*          hn    = (float*)alloc((size_t)N * HID * 4);
    float*          coord = (float*)alloc((size_t)N * 3 * 4);
    float*          csum  = (float*)alloc((size_t)N * 3 * 4);
    unsigned short* eattr = (unsigned short*)alloc((size_t)E * 64 * 2);
    float*          geom  = (float*)alloc((size_t)E * 10 * 4);
    unsigned short* ef    = (unsigned short*)alloc((size_t)E * HID * 2);
    unsigned* w1p[4]; unsigned* w2p[4]; unsigned* wn1p[4]; unsigned* wn2p[4]; unsigned* wc1p[4];
    for (int l = 0; l < 4; ++l) {
        w1p[l]  = (unsigned*)alloc((size_t)176 * 128 * 4);
        w2p[l]  = (unsigned*)alloc((size_t)64 * 128 * 4);
        wn1p[l] = (unsigned*)alloc((size_t)128 * 128 * 4);
        wn2p[l] = (unsigned*)alloc((size_t)64 * 128 * 4);
        wc1p[l] = (unsigned*)alloc((size_t)64 * 128 * 4);
    }
    (void)ws_size; (void)n_in; (void)out_size;

    // pack weights -> bf16, K-pair interleaved, fragment-major chunks
    for (int l = 0; l < 4; ++l) {
        const int base = 12 + l * 14;
        const float* e1W = (const float*)d_in[base + 0];
        const float* e2W = (const float*)d_in[base + 2];
        const float* n1W = (const float*)d_in[base + 4];
        const float* n2W = (const float*)d_in[base + 6];
        const float* c1W = (const float*)d_in[base + 8];
        int tp;
        tp = 176 * 128; pack_weights_e1_kernel<<<(tp + 255) / 256, 256, 0, stream>>>(e1W, w1p[l]);
        tp = 64 * 128;  pack_weights_kernel<<<(tp + 255) / 256, 256, 0, stream>>>(e2W, w2p[l], 128, 128, 128);
        tp = 128 * 128; pack_weights_kernel<<<(tp + 255) / 256, 256, 0, stream>>>(n1W, wn1p[l], 256, 256, 128);
        tp = 64 * 128;  pack_weights_kernel<<<(tp + 255) / 256, 256, 0, stream>>>(n2W, wn2p[l], 128, 128, 128);
        tp = 64 * 128;  pack_weights_kernel<<<(tp + 255) / 256, 256, 0, stream>>>(c1W, wc1p[l], 128, 128, 128);
    }

    // h = emb_in(h_in)
    emb_in_kernel<<<(N * HID + 255) / 256, 256, 0, stream>>>(h_in, embin_W, embin_b, h, h_bf, N);

    // centroid + coord
    zero_kernel<<<(N * 3 + 255) / 256, 256, 0, stream>>>(csum, (size_t)N * 3);
    centroid_kernel<<<(N * 3 + 255) / 256, 256, 0, stream>>>(x, nn_ptr, csum, N);
    coord_init_kernel<<<(N * 3 + 255) / 256, 256, 0, stream>>>(x, nn_ptr, csum, coord, N);

    // initial frame + fused edge feature MLP
    frame_kernel<<<(E + 255) / 256, 256, 0, stream>>>(row, col, coord, geom, E);
    edge_attr_kernel<<<(E + 127) / 128, 128, 0, stream>>>(row, col, coord, geom,
                                                          f1W, f1b, f2W, f2b, eattr, E);

    const int eBlocks = (E + 63) / 64;
    const int nBlocks = (N + 63) / 64;
    for (int l = 0; l < 4; ++l) {
        const int base = 12 + l * 14;
        const float* e1b = (const float*)d_in[base + 1];
        const float* e2b = (const float*)d_in[base + 3];
        const float* n1b = (const float*)d_in[base + 5];
        const float* n2b = (const float*)d_in[base + 7];
        const float* c1b = (const float*)d_in[base + 9];
        const float* c2W = (const float*)d_in[base + 10];
        const float* c2b = (const float*)d_in[base + 11];
        const float* lng = (const float*)d_in[base + 12];
        const float* lnb = (const float*)d_in[base + 13];

        frame_kernel<<<(E + 255) / 256, 256, 0, stream>>>(row, col, coord, geom, E);
        edge_mlp_kernel<<<eBlocks, 128, 0, stream>>>(row, col, h_bf, eattr, geom,
                                                     w1p[l], e1b, w2p[l], e2b, ef, E);
        zero_kernel<<<(int)(((size_t)N * HID + 255) / 256), 256, 0, stream>>>(aggb, (size_t)N * HID);
        agg_kernel<<<(int)(((size_t)E * 32 + 255) / 256), 256, 0, stream>>>(col, ef, aggb, E);
        node_mlp_kernel<<<nBlocks, 128, 0, stream>>>(h_bf, aggb, wn1p[l], n1b, wn2p[l], n2b, hn, N);
        layernorm_kernel<<<(N + 127) / 128, 128, 0, stream>>>(hn, lng, lnb, h, h_bf, N);
        coord_update_kernel<<<eBlocks, 128, 0, stream>>>(row, ef, wc1p[l], c1b, c2W, c2b,
                                                         geom, coord, E);
    }

    emb_out_kernel<<<(N + 127) / 128, 128, 0, stream>>>(h, embout_W, embout_b,
                                                        (float*)d_out, N);
    xout_kernel<<<(N * 3 + 255) / 256, 256, 0, stream>>>(coord, csum, nn_ptr,
                                                         (float*)d_out + (size_t)N * 16, N);
}
